// ODEBlock_79688823210325
// MI455X (gfx1250) — matrix operand order repaired and checked by reading, not compiled, not run on hardware
//
#include <hip/hip_runtime.h>

typedef __attribute__((ext_vector_type(16))) _Float16 v16h;
typedef __attribute__((ext_vector_type(8)))  float    v8f;
typedef __attribute__((ext_vector_type(4)))  float    vf4;

#ifndef __has_builtin
#define __has_builtin(x) 0
#endif

#if __has_builtin(__builtin_amdgcn_exp2f)
#define FAST_EXP2(x) __builtin_amdgcn_exp2f(x)
#else
#define FAST_EXP2(x) exp2f(x)
#endif
#if __has_builtin(__builtin_amdgcn_rcpf)
#define FAST_RCP(x) __builtin_amdgcn_rcpf(x)
#else
#define FAST_RCP(x) (1.0f / (x))
#endif

__device__ __forceinline__ float tanh_sw(float v) {
  float a = __builtin_fabsf(v);
  float e = FAST_EXP2(a * -2.8853900817779268f);   // exp(-2a), in (0,1]
  float r = (1.0f - e) * FAST_RCP(1.0f + e);
  return __builtin_copysignf(r, v);
}

// CDNA5 hardware transcendental v_tanh_f32
#if __has_builtin(__builtin_amdgcn_tanhf)
#define TANH_FAST(x) __builtin_amdgcn_tanhf(x)
#elif __has_builtin(__builtin_amdgcn_tanh_f32)
#define TANH_FAST(x) __builtin_amdgcn_tanh_f32(x)
#else
#define TANH_FAST(x) tanh_sw(x)
#endif

#define B_SZ   16384
#define DDIM   32
#define HDIM   64
#define TSTEPS 50
#define NSUB   2
#define WAVES_PER_BLOCK 8
#define RSTRIDE 36   // dwords per staging row (32 + 4 pad): 16B aligned

#define WMMA_F16(a, b, c) \
  __builtin_amdgcn_wmma_f32_16x16x32_f16(false, (a), false, (b), (short)0, (c), false, false)

// Compiler-only memory fence; hardware DS from one wave is in-order, and the
// compiler inserts precise s_wait_dscnt before each load's use.
#define LDS_FENCE() asm volatile("" ::: "memory")

__global__ __launch_bounds__(256)
void node_rk4_wmma(const float* __restrict__ x0,
                   const float* __restrict__ W1,
                   const float* __restrict__ b1,
                   const float* __restrict__ W2,
                   const float* __restrict__ b2,
                   const float* __restrict__ times,
                   const int*   __restrict__ ids,
                   float* __restrict__ out)
{
  __shared__ __align__(16) float stage[WAVES_PER_BLOCK][16 * RSTRIDE];
  const int lane = threadIdx.x & 31;
  const int wave = threadIdx.x >> 5;
  const int n    = lane & 15;       // column index inside a 16-wide tile
  const int hi   = lane >> 4;       // 0: lanes 0-15, 1: lanes 16-31
  const int m0   = (blockIdx.x * WAVES_PER_BLOCK + wave) << 4;  // first batch row
  float* const st = &stage[wave][0];

  __builtin_prefetch(W1, 0, 3);
  __builtin_prefetch(W2, 0, 3);

  const int klo = hi << 3;   // A-operand K offset for this lane half
  const int kbB = hi << 4;   // B-operand K offset for this lane half

  // ---- gemm1' (h^T = W1^T @ z^T): A = W1^T, 4 tiles (M=h 16u.., K=d 0..31), f16
  v16h w1a[4];
#pragma unroll
  for (int u = 0; u < 4; ++u)
#pragma unroll
    for (int e = 0; e < 8; ++e) {
      w1a[u][e]     = (_Float16)W1[(kbB + e) * HDIM + 16 * u + n];
      w1a[u][8 + e] = (_Float16)W1[(kbB + 8 + e) * HDIM + 16 * u + n];
    }
  // ---- gemm2 (y = h @ W2): B = W2, K chunk c (2x32), N=d tile t (2), f16
  v16h w2b[2][2];
#pragma unroll
  for (int c = 0; c < 2; ++c)
#pragma unroll
    for (int t = 0; t < 2; ++t)
#pragma unroll
      for (int e = 0; e < 16; ++e)
        w2b[c][t][e] = (_Float16)W2[(32 * c + klo + (e & 7) + ((e >> 3) << 4)) * DDIM + 16 * t + n];

  // biases pre-laid in the WMMA C layouts, fed directly as the C operand
  v8f b1c[4];                      // gemm1' C: M = h = 16u + v + 8*hi (VGPR-major)
#pragma unroll
  for (int u = 0; u < 4; ++u)
#pragma unroll
    for (int v = 0; v < 8; ++v)
      b1c[u][v] = b1[16 * u + 8 * hi + v];
  v8f b2c[2];                      // gemm2 C: N = d = 16t + n (lane-major)
#pragma unroll
  for (int t = 0; t < 2; ++t)
#pragma unroll
    for (int v = 0; v < 8; ++v)
      b2c[t][v] = b2[16 * t + n];

  // ---- x state in gemm2-C layout: tile t: N=d=16t+n (lanes), M=batch=v+8hi (VGPRs)
  v8f xs[2];
#pragma unroll
  for (int t = 0; t < 2; ++t)
#pragma unroll
    for (int v = 0; v < 8; ++v)
      xs[t][v] = x0[(size_t)(m0 + v + 8 * hi) * DDIM + 16 * t + n];

  const int dsel = ids[m0 + n];
  if (lane < 16)
    out[m0 + n] = x0[(size_t)(m0 + n) * DDIM + dsel];   // t = 0 output

  // y = tanh((z + s*k) @ W1 + b1) @ W2 + b2 ; z,k,y all in gemm2-C layout.
  // Single LDS turnaround per eval: only z^T needs a real transpose; h^T's
  // C layout already matches gemm2's A layout lane-for-lane (register repack).
  auto odefunc = [&](v8f y[2], const v8f z[2], const v8f k[2], float s) {
    LDS_FENCE();
#pragma unroll
    for (int t = 0; t < 2; ++t)
#pragma unroll
      for (int v = 0; v < 8; ++v)
        st[(v + 8 * hi) * RSTRIDE + 16 * t + n] = z[t][v] + s * k[t][v];
    LDS_FENCE();
    // B operand of gemm1': z^T (N=batch on lanes, K=d on VGPRs)
    const float* r = st + n * RSTRIDE + kbB;
    vf4 q0 = *(const vf4*)(r);
    vf4 q1 = *(const vf4*)(r + 4);
    vf4 q2 = *(const vf4*)(r + 8);
    vf4 q3 = *(const vf4*)(r + 12);
    v16h bz;
#pragma unroll
    for (int i2 = 0; i2 < 4; ++i2) {
      bz[i2]      = (_Float16)q0[i2];
      bz[4 + i2]  = (_Float16)q1[i2];
      bz[8 + i2]  = (_Float16)q2[i2];
      bz[12 + i2] = (_Float16)q3[i2];
    }
    // gemm1': four back-to-back WMMAs, bias fed as the C operand
    v8f hm[4];
#pragma unroll
    for (int u = 0; u < 4; ++u)
      hm[u] = WMMA_F16(w1a[u], bz, b1c[u]);
    // hardware tanh pass
#pragma unroll
    for (int u = 0; u < 4; ++u)
#pragma unroll
      for (int v = 0; v < 8; ++v)
        hm[u][v] = TANH_FAST(hm[u][v]);
    // pack gemm2 A operands in-register (no LDS, no cross-lane movement)
    v16h ah[2];
#pragma unroll
    for (int c = 0; c < 2; ++c)
#pragma unroll
      for (int e = 0; e < 8; ++e) {
        ah[c][e]     = (_Float16)hm[2 * c][e];
        ah[c][8 + e] = (_Float16)hm[2 * c + 1][e];
      }
    // gemm2: two independent accumulation chains, interleaved
    v8f c0 = WMMA_F16(ah[0], w2b[0][0], b2c[0]);
    v8f c1 = WMMA_F16(ah[0], w2b[0][1], b2c[1]);
    c0 = WMMA_F16(ah[1], w2b[1][0], c0);
    c1 = WMMA_F16(ah[1], w2b[1][1], c1);
    y[0] = c0;
    y[1] = c1;
  };

  v8f ya[2], yb[2], acc[2];
#pragma unroll 1
  for (int i = 0; i < TSTEPS - 1; ++i) {
    const float dt = (times[i + 1] - times[i]) * (1.0f / NSUB);
#pragma unroll 1
    for (int sstep = 0; sstep < NSUB; ++sstep) {
      odefunc(ya, xs, xs, 0.0f);                    // k1
#pragma unroll
      for (int t = 0; t < 2; ++t) acc[t] = ya[t];
      odefunc(yb, xs, ya, 0.5f * dt);               // k2
#pragma unroll
      for (int t = 0; t < 2; ++t)
#pragma unroll
        for (int v = 0; v < 8; ++v) acc[t][v] += 2.0f * yb[t][v];
      odefunc(ya, xs, yb, 0.5f * dt);               // k3
#pragma unroll
      for (int t = 0; t < 2; ++t)
#pragma unroll
        for (int v = 0; v < 8; ++v) acc[t][v] += 2.0f * ya[t][v];
      odefunc(yb, xs, ya, dt);                      // k4
      const float c6 = dt * (1.0f / 6.0f);
#pragma unroll
      for (int t = 0; t < 2; ++t)
#pragma unroll
        for (int v = 0; v < 8; ++v)
          xs[t][v] += c6 * (acc[t][v] + yb[t][v]);
    }
    // gather-output through the staging tile
    LDS_FENCE();
#pragma unroll
    for (int t = 0; t < 2; ++t)
#pragma unroll
      for (int v = 0; v < 8; ++v)
        st[(v + 8 * hi) * RSTRIDE + 16 * t + n] = xs[t][v];
    LDS_FENCE();
    if (lane < 16)
      out[(size_t)(i + 1) * B_SZ + m0 + n] = st[n * RSTRIDE + dsel];
    LDS_FENCE();
  }
}

extern "C" void kernel_launch(void* const* d_in, const int* in_sizes, int n_in,
                              void* d_out, int out_size, void* d_ws, size_t ws_size,
                              hipStream_t stream) {
  (void)in_sizes; (void)n_in; (void)d_ws; (void)ws_size; (void)out_size;
  const float* x     = (const float*)d_in[0];
  const float* W1    = (const float*)d_in[1];
  const float* b1    = (const float*)d_in[2];
  const float* W2    = (const float*)d_in[3];
  const float* b2    = (const float*)d_in[4];
  const float* times = (const float*)d_in[5];
  const int*   ids   = (const int*)d_in[6];
  float* out = (float*)d_out;

  dim3 grid(B_SZ / (16 * WAVES_PER_BLOCK));   // 128 blocks x 8 waves = 1024 tiles
  node_rk4_wmma<<<grid, 256, 0, stream>>>(x, W1, b1, W2, b2, times, ids, out);
}
